// moe_AgentAttention_25383256720181
// MI455X (gfx1250) — compile-verified, matrix operand order checked by reading, not run on hardware
//
#include <hip/hip_runtime.h>
#include <hip/hip_bf16.h>
#include <math.h>

#define BB   4
#define NN   8192
#define DIMM 512
#define HH   8
#define DHH  64
#define MM   256
#define QK_SCALE 0.125f                  // 64^-0.5
#define LOG2E 1.44269504088896340736f

typedef __attribute__((ext_vector_type(16))) __bf16 v16bf;
typedef __attribute__((ext_vector_type(8)))  float  v8f;
typedef __attribute__((ext_vector_type(2)))  unsigned u32x2;
typedef __attribute__((ext_vector_type(4)))  unsigned u32x4;

union BfVec { v16bf v; unsigned short s[16]; unsigned u[8]; u32x4 q[2]; };

// base-2 exponential -> bare v_exp_f32
__device__ __forceinline__ float fast_exp2(float x) {
#if __has_builtin(__builtin_amdgcn_exp2f)
  return __builtin_amdgcn_exp2f(x);
#else
  return exp2f(x);
#endif
}

// two f32 -> packed bf16x2 in ONE v_perm_b32 (truncate)
__device__ __forceinline__ unsigned pkbf2(float lo, float hi) {
  union { float f; unsigned u; } a, b;
  a.f = lo; b.f = hi;
  return __builtin_amdgcn_perm(b.u, a.u, 0x07060302u);
}

__device__ __forceinline__ unsigned short f2bf(float f) {
  union { float f; unsigned u; } c; c.f = f;
  return (unsigned short)(c.u >> 16);
}

__device__ __forceinline__ v8f wmma_bf16(v16bf a, v16bf b, v8f c) {
  return __builtin_amdgcn_wmma_f32_16x16x32_bf16(false, a, false, b, (short)0, c,
                                                 false, false);
}

// gfx1250 16-bit A/B operand: lane<16 holds K{0..7,16..23}, lane>=16 holds
// K{8..15,24..31} => two contiguous 16B runs. element(row,k)=p[row*ld+k].
__device__ __forceinline__ v16bf load_opmajor(const unsigned short* p, int ld,
                                              int row, int kbase, int lane) {
  const int hf = (lane >> 4) & 1;
  const unsigned short* rp = p + row * ld + kbase + hf * 8;
  BfVec r;
  r.q[0] = *(const u32x4*)(rp);
  r.q[1] = *(const u32x4*)(rp + 16);
  return r.v;
}

// pack two exponentiated C-layout tiles (rows = K) directly into a B operand
__device__ __forceinline__ v16bf pack_b_from_acc(v8f t0, v8f t1) {
  BfVec r;
#pragma unroll
  for (int j = 0; j < 4; ++j) {
    r.u[j]     = pkbf2(t0[2 * j], t0[2 * j + 1]);
    r.u[4 + j] = pkbf2(t1[2 * j], t1[2 * j + 1]);
  }
  return r.v;
}

// CDNA5 async global->LDS copy: each lane moves 16B; tracked by ASYNCcnt.
__device__ __forceinline__ void async_load_b128(unsigned lds_off,
                                                const unsigned short* base,
                                                unsigned goff) {
  asm volatile("global_load_async_to_lds_b128 %0, %1, %2 offset:0"
               :: "v"(lds_off), "v"(goff), "s"(base)
               : "memory");
}

// ---------------------------------------------------------------------------
// Kernel 0a: x f32 -> bf16 (row-major), packed
// ---------------------------------------------------------------------------
__global__ void cvt_x_kernel(const float* __restrict__ a,
                             unsigned short* __restrict__ o, int n4) {
  int i = blockIdx.x * blockDim.x + threadIdx.x;
  if (i < n4) {
    float4 f = *(const float4*)(a + 4 * (size_t)i);
    u32x2 pk; pk[0] = pkbf2(f.x, f.y); pk[1] = pkbf2(f.z, f.w);
    *(u32x2*)(o + 4 * (size_t)i) = pk;
  }
}

// ---------------------------------------------------------------------------
// Kernel 0b: W f32 [512][1536] -> W^T bf16 [1536][512]
// ---------------------------------------------------------------------------
__global__ void cvt_wT_kernel(const float* __restrict__ W,
                              unsigned short* __restrict__ Wt) {
  int id = blockIdx.x * 256 + threadIdx.x;   // 1536*256 ids (one bf16 pair each)
  int c = id >> 8, kp = id & 255;
  float a = W[(size_t)(2 * kp)     * (3 * DIMM) + c];
  float b = W[(size_t)(2 * kp + 1) * (3 * DIMM) + c];
  ((unsigned*)Wt)[(size_t)c * 256 + kp] = pkbf2(a, b);
}

// ---------------------------------------------------------------------------
// Kernel 0c: agent f32 -> bf16 scaled by log2(e) (feeds both logit GEMMs,
// so all softmax exps become bare v_exp_f32 via exp2)
// ---------------------------------------------------------------------------
__global__ void cvt_agent_kernel(const float* __restrict__ a,
                                 unsigned short* __restrict__ abf, int n) {
  int i = blockIdx.x * blockDim.x + threadIdx.x;
  if (2 * i < n) {
    float2 f = *(const float2*)(a + 2 * i);
    ((unsigned*)abf)[i] = pkbf2(f.x * LOG2E, f.y * LOG2E);
  }
}

// ---------------------------------------------------------------------------
// Kernel 1: qkv = x @ W (bf16 WMMA). Tiles double-buffered in LDS via
// global_load_async_to_lds_b128 (ASYNCcnt), zero conversion VALU in the loop.
// q/k row-major [b][h][n][dh]; v TRANSPOSED [b][h][d][n]; q pre-scaled.
// ---------------------------------------------------------------------------
__device__ __forceinline__ void issue_tile_loads(
    const unsigned short* xbf, const unsigned short* wt,
    unsigned short* xbuf, unsigned short* wbuf,
    int gr0, int gc0, int kk0, int t) {
  // x tile 128x32 bf16 = 512 16B-chunks, 2 per lane
#pragma unroll
  for (int j = 0; j < 2; ++j) {
    int c = t * 2 + j;
    int r = c >> 2, part = c & 3;
    unsigned loff = (unsigned)(size_t)(&xbuf[r * 40 + part * 8]);
    unsigned goff = (unsigned)(((gr0 + r) * DIMM + kk0 + part * 8) * 2);
    async_load_b128(loff, xbf, goff);
  }
  // W^T tile 64x32 bf16 = 256 chunks, 1 per lane
  {
    int c = t >> 2, part = t & 3;
    unsigned loff = (unsigned)(size_t)(&wbuf[c * 40 + part * 8]);
    unsigned goff = (unsigned)(((gc0 + c) * DIMM + kk0 + part * 8) * 2);
    async_load_b128(loff, wt, goff);
  }
}

__global__ __launch_bounds__(256)
void qkv_gemm_kernel(const unsigned short* __restrict__ xbf,
                     const unsigned short* __restrict__ wt,
                     unsigned short* __restrict__ qbf,
                     unsigned short* __restrict__ kbf,
                     unsigned short* __restrict__ vbf) {
  __shared__ unsigned short xs0[128 * 40];
  __shared__ unsigned short xs1[128 * 40];
  __shared__ unsigned short ws0[64 * 40];
  __shared__ unsigned short ws1[64 * 40];
  __shared__ unsigned short outs[8][16 * 64];
  const int t = threadIdx.x, lane = t & 31, wave = t >> 5;
  const int hf = lane >> 4;
  const int gr0 = blockIdx.x * 128;
  const int gc0 = blockIdx.y * 64;

  v8f acc[4] = {};

  issue_tile_loads(xbf, wt, xs0, ws0, gr0, gc0, 0, t);

  for (int kt = 0; kt < 16; ++kt) {
    unsigned short* xcur = (kt & 1) ? xs1 : xs0;
    unsigned short* wcur = (kt & 1) ? ws1 : ws0;
    if (kt < 15) {
      unsigned short* xnxt = (kt & 1) ? xs0 : xs1;
      unsigned short* wnxt = (kt & 1) ? ws0 : ws1;
      issue_tile_loads(xbf, wt, xnxt, wnxt, gr0, gc0, (kt + 1) * 32, t);
      asm volatile("s_wait_asynccnt 3" ::: "memory");   // current tile landed
    } else {
      asm volatile("s_wait_asynccnt 0" ::: "memory");
    }
    __syncthreads();
    v16bf A = load_opmajor(xcur, 40, wave * 16 + (lane & 15), 0, lane);
#pragma unroll
    for (int ct = 0; ct < 4; ++ct) {
      v16bf Bm = load_opmajor(wcur, 40, ct * 16 + (lane & 15), 0, lane);
      acc[ct] = wmma_bf16(A, Bm, acc[ct]);
    }
    __syncthreads();   // all reads done before buffer is overwritten
  }

  const int which = gc0 >> 9;            // block-uniform: 0=q 1=k 2=v
  const int h = (gc0 & 511) >> 6;
  const int b = gr0 >> 13;
  const int nb = (gr0 & (NN - 1)) + wave * 16;

  if (which == 2) {
    unsigned short* vT = vbf + (size_t)(b * HH + h) * DHH * NN;
#pragma unroll
    for (int ct = 0; ct < 4; ++ct) {
      int d = ct * 16 + (lane & 15);
      u32x4 pk;
      pk[0] = pkbf2(acc[ct][0], acc[ct][1]);
      pk[1] = pkbf2(acc[ct][2], acc[ct][3]);
      pk[2] = pkbf2(acc[ct][4], acc[ct][5]);
      pk[3] = pkbf2(acc[ct][6], acc[ct][7]);
      *(u32x4*)(&vT[(size_t)d * NN + nb + 8 * hf]) = pk;
    }
  } else {
    unsigned short* dstbuf = (which == 0) ? qbf : kbf;
    const float sc = (which == 0) ? QK_SCALE : 1.0f;
#pragma unroll
    for (int ct = 0; ct < 4; ++ct)
#pragma unroll
      for (int i = 0; i < 8; ++i)
        outs[wave][(i + 8 * hf) * 64 + ct * 16 + (lane & 15)] = f2bf(acc[ct][i] * sc);
    __asm__ volatile("s_wait_dscnt 0" ::: "memory");
    unsigned short* dst = dstbuf + ((size_t)(b * HH + h) * NN + nb) * DHH;
#pragma unroll
    for (int j = 0; j < 4; ++j) {
      int e = (j * 32 + lane) * 8;
      *(u32x4*)(&dst[e]) = *(const u32x4*)(&outs[wave][e]);
    }
  }
}

// ---------------------------------------------------------------------------
// Kernel 3: kvT[b,h,d,m] = softmax_dh( softmax_n(agent @ k^T) @ v )^T
// S^T = k @ agent^T: n-softmax is per-lane; exp2 registers ARE the next
// WMMA's B operand. Max-subtraction dropped (identity; logits sigma~8).
// ---------------------------------------------------------------------------
__global__ __launch_bounds__(256)
void kv_kernel(const unsigned short* __restrict__ kbf,
               const unsigned short* __restrict__ vT,
               const unsigned short* __restrict__ agbf,
               unsigned short* __restrict__ kvT) {
  __shared__ float kvpart[8][16][64];   // [wave][m][d]
  __shared__ float rsp[8][16];
  __shared__ float kvfin[16][64];

  const int t = threadIdx.x, lane = t & 31, wave = t >> 5;
  const int hf = lane >> 4;
  const int bh = blockIdx.x;
  const int h = bh & (HH - 1);
  const int m0 = blockIdx.y * 16;

  const unsigned short* kb  = kbf + (size_t)bh * NN * DHH;
  const unsigned short* vTb = vT  + (size_t)bh * DHH * NN;
  const unsigned short* ag  = agbf + ((size_t)h * MM + m0) * DHH;

  v16bf Bag0 = load_opmajor(ag, DHH, lane & 15, 0, lane);
  v16bf Bag1 = load_opmajor(ag, DHH, lane & 15, 32, lane);

  v8f kvaccT[4] = {};
  float rs = 0.f;

  for (int n0 = wave * 32; n0 < NN; n0 += 256) {
    v8f z = {};
    v16bf Ak00 = load_opmajor(kb + (size_t)n0 * DHH, DHH, lane & 15, 0, lane);
    v16bf Ak01 = load_opmajor(kb + (size_t)n0 * DHH, DHH, lane & 15, 32, lane);
    v8f T0 = wmma_bf16(Ak01, Bag1, wmma_bf16(Ak00, Bag0, z));
    v16bf Ak10 = load_opmajor(kb + (size_t)(n0 + 16) * DHH, DHH, lane & 15, 0, lane);
    v16bf Ak11 = load_opmajor(kb + (size_t)(n0 + 16) * DHH, DHH, lane & 15, 32, lane);
    v8f T1 = wmma_bf16(Ak11, Bag1, wmma_bf16(Ak10, Bag0, z));

#pragma unroll
    for (int j = 0; j < 8; ++j) {
      T0[j] = fast_exp2(T0[j]);      // log2e folded into agent
      T1[j] = fast_exp2(T1[j]);
      rs += T0[j] + T1[j];
    }
    v16bf Pb = pack_b_from_acc(T0, T1);

#pragma unroll
    for (int dt = 0; dt < 4; ++dt) {
      v16bf Av = load_opmajor(vTb, NN, dt * 16 + (lane & 15), n0, lane);
      kvaccT[dt] = wmma_bf16(Av, Pb, kvaccT[dt]);
    }
  }

  rs += __shfl_xor(rs, 16, 32);
  if (lane < 16) rsp[wave][lane] = rs;

  const int m = lane & 15;
#pragma unroll
  for (int dt = 0; dt < 4; ++dt) {
    int d0 = dt * 16 + 8 * hf;
    float4 lo = make_float4(kvaccT[dt][0], kvaccT[dt][1], kvaccT[dt][2], kvaccT[dt][3]);
    float4 hi = make_float4(kvaccT[dt][4], kvaccT[dt][5], kvaccT[dt][6], kvaccT[dt][7]);
    *(float4*)(&kvpart[wave][m][d0])     = lo;
    *(float4*)(&kvpart[wave][m][d0 + 4]) = hi;
  }
  __syncthreads();

#pragma unroll
  for (int j = 0; j < 4; ++j) {
    int e = t + j * 256;
    int row = e >> 6, col = e & 63;
    float val = 0.f, gs = 0.f;
#pragma unroll
    for (int w = 0; w < 8; ++w) {
      val += kvpart[w][row][col];
      gs  += rsp[w][row];
    }
    kvfin[row][col] = val / gs;
  }
  __syncthreads();

  if (t < 16) {
    int row = t;
    float mx = -3.0e38f;
    for (int d = 0; d < DHH; ++d) mx = fmaxf(mx, kvfin[row][d]);
    float s = 0.f;
    for (int d = 0; d < DHH; ++d) s += fast_exp2((kvfin[row][d] - mx) * LOG2E);
    float inv = 1.0f / s;
    for (int d = 0; d < DHH; ++d)
      kvT[((size_t)bh * DHH + d) * MM + m0 + row] =
          f2bf(fast_exp2((kvfin[row][d] - mx) * LOG2E) * inv);
  }
}

// ---------------------------------------------------------------------------
// Kernel 4: out = softmax_m(q_scaled @ agent^T) @ kv, all-transposed form,
// no LDS, exp2 registers reused as B operands, packed float4 stores.
// ---------------------------------------------------------------------------
__global__ __launch_bounds__(256)
void out_kernel(const unsigned short* __restrict__ qbf,
                const unsigned short* __restrict__ agbf,
                const unsigned short* __restrict__ kvT,
                float* __restrict__ out) {
  const int t = threadIdx.x, lane = t & 31, wave = t >> 5;
  const int hf = lane >> 4;
  const int bh = blockIdx.x;
  const int b = bh >> 3, h = bh & (HH - 1);
  const int n0 = blockIdx.y * 128 + wave * 16;

  const unsigned short* qb = qbf + ((size_t)bh * NN + n0) * DHH;
  const unsigned short* ag = agbf + (size_t)h * MM * DHH;
  const unsigned short* kv = kvT + (size_t)bh * DHH * MM;

  v16bf Bq0 = load_opmajor(qb, DHH, lane & 15, 0, lane);
  v16bf Bq1 = load_opmajor(qb, DHH, lane & 15, 32, lane);

  v8f Lt[16];
  float s = 0.f;
#pragma unroll
  for (int mt = 0; mt < 16; ++mt) {
    v8f z = {};
    v16bf A0 = load_opmajor(ag + (size_t)mt * 16 * DHH, DHH, lane & 15, 0, lane);
    v16bf A1 = load_opmajor(ag + (size_t)mt * 16 * DHH, DHH, lane & 15, 32, lane);
    Lt[mt] = wmma_bf16(A1, Bq1, wmma_bf16(A0, Bq0, z));
#pragma unroll
    for (int j = 0; j < 8; ++j) {
      Lt[mt][j] = fast_exp2(Lt[mt][j]);   // log2e folded into agent
      s += Lt[mt][j];
    }
  }
  s += __shfl_xor(s, 16, 32);
  const float oinv = 1.0f / s;

  v8f oaccT[4] = {};
#pragma unroll
  for (int mc = 0; mc < 8; ++mc) {
    v16bf Pb = pack_b_from_acc(Lt[2 * mc], Lt[2 * mc + 1]);
#pragma unroll
    for (int dt = 0; dt < 4; ++dt) {
      v16bf Akv = load_opmajor(kv, MM, dt * 16 + (lane & 15), mc * 32, lane);
      oaccT[dt] = wmma_bf16(Akv, Pb, oaccT[dt]);
    }
  }

  const int n = n0 + (lane & 15);
  float* op = out + ((size_t)b * NN + n) * DIMM + h * DHH;
#pragma unroll
  for (int dt = 0; dt < 4; ++dt) {
    int d0 = dt * 16 + 8 * hf;
    float4 lo = make_float4(oaccT[dt][0] * oinv, oaccT[dt][1] * oinv,
                            oaccT[dt][2] * oinv, oaccT[dt][3] * oinv);
    float4 hi = make_float4(oaccT[dt][4] * oinv, oaccT[dt][5] * oinv,
                            oaccT[dt][6] * oinv, oaccT[dt][7] * oinv);
    *(float4*)(op + d0)     = lo;
    *(float4*)(op + d0 + 4) = hi;
  }
}

// ---------------------------------------------------------------------------
extern "C" void kernel_launch(void* const* d_in, const int* in_sizes, int n_in,
                              void* d_out, int out_size, void* d_ws, size_t ws_size,
                              hipStream_t stream) {
  const float* x     = (const float*)d_in[0];
  const float* Wqkv  = (const float*)d_in[1];
  const float* agent = (const float*)d_in[2];
  float* out = (float*)d_out;

  unsigned short* ws = (unsigned short*)d_ws;
  const size_t QE = (size_t)BB * HH * NN * DHH;   // 16,777,216 (== B*N*DIM)
  const size_t WT = (size_t)(3 * DIMM) * DIMM;    // 786,432
  const size_t AE = (size_t)HH * MM * DHH;        // 131,072
  unsigned short* xbf  = ws;                 // [b*n][dim]      bf16
  unsigned short* wtbf = xbf + QE;           // [1536][512]     bf16 (W^T)
  unsigned short* qbf  = wtbf + WT;          // [b][h][n][dh]
  unsigned short* kbf  = qbf + QE;           // [b][h][n][dh]
  unsigned short* vT   = kbf + QE;           // [b][h][d][n]
  unsigned short* agbf = vT + QE;            // [h][m][dh] * log2e
  unsigned short* kvT  = agbf + AE;          // [b][h][d][m]

  cvt_x_kernel<<<(int)(QE / 4 / 256), 256, 0, stream>>>(x, xbf, (int)(QE / 4));
  cvt_wT_kernel<<<(3 * DIMM), 256, 0, stream>>>(Wqkv, wtbf);
  cvt_agent_kernel<<<(int)((AE / 2 + 255) / 256), 256, 0, stream>>>(agent, agbf, (int)AE);

  dim3 g1((BB * NN) / 128, (3 * DIMM) / 64);      // 256 x 24
  qkv_gemm_kernel<<<g1, 256, 0, stream>>>(xbf, wtbf, qbf, kbf, vT);

  dim3 g3(BB * HH, MM / 16);                      // 32 x 16
  kv_kernel<<<g3, 256, 0, stream>>>(kbf, vT, agbf, kvT);

  dim3 g4(BB * HH, NN / 128);                     // 32 x 64
  out_kernel<<<g4, 256, 0, stream>>>(qbf, agbf, kvT, out);
}
